// DPNModel_47132971106930
// MI455X (gfx1250) — matrix operand order repaired and checked by reading, not compiled, not run on hardware
//
#include <hip/hip_runtime.h>
#include <math.h>

typedef __attribute__((ext_vector_type(16))) _Float16 v16h;
typedef __attribute__((ext_vector_type(8)))  float    v8f;

#define WAVE 32

// ---------------- elementwise helpers ----------------
__global__ void fill_bias_kernel(float* __restrict__ out, const float* __restrict__ bias,
                                 int M, int Npix) {
  int idx = blockIdx.x * blockDim.x + threadIdx.x;
  if (idx < M * Npix) out[idx] = bias[idx / Npix];
}

__global__ void relu_kernel(float* __restrict__ x, int n) {
  int idx = blockIdx.x * blockDim.x + threadIdx.x;
  if (idx < n) x[idx] = fmaxf(x[idx], 0.f);
}

// ---------------- direct fp32 conv (tiny layers: conv0 Cin=3, conv15 Cout=21) --------
__global__ void conv_direct_kernel(const float* __restrict__ in, const float* __restrict__ w,
                                   const float* __restrict__ bias, float* __restrict__ out,
                                   int Cin, int H, int W, int Cout, int K,
                                   int pad, int dil, int OH, int OW, int relu) {
  int idx = blockIdx.x * blockDim.x + threadIdx.x;
  int total = Cout * OH * OW;
  if (idx >= total) return;
  int o  = idx / (OH * OW);
  int r  = idx - o * OH * OW;
  int oy = r / OW, ox = r - oy * OW;
  float acc = bias[o];
  for (int ci = 0; ci < Cin; ++ci)
    for (int ky = 0; ky < K; ++ky) {
      int iy = oy + ky * dil - pad;
      if (iy < 0 || iy >= H) continue;
      for (int kx = 0; kx < K; ++kx) {
        int ix = ox + kx * dil - pad;
        if (ix < 0 || ix >= W) continue;
        acc += in[(ci * H + iy) * W + ix] * w[((o * Cin + ci) * K + ky) * K + kx];
      }
    }
  out[idx] = relu ? fmaxf(acc, 0.f) : acc;
}

// ---------------- WMMA implicit-GEMM conv (f16 inputs, f32 accumulate) ----------------
// C[Cout, OH*OW] = bias + W[Cout, Cin*K*K] * im2col(X)[Cin*K*K, OH*OW]
// One wave computes a 16(M) x 64(N) strip: A fragment loaded ONCE per K-step
// (4x global_load_b128 per lane) and reused for 4 WMMAs -> weights stream from HBM in a
// single pass even when N is small (conv13: 411 MB read exactly once). B fragments are
// built one tile at a time (single live fragment keeps register pressure low so the
// 16 clamp-select gather loads batch, and the gather of tile t+1 overlaps WMMA(t));
// WMMAs are unconditional (invalid tiles gather zeros). Full K-steps skip the per-
// element k-bound test (chunk boundaries are multiples of 32). K is split across
// blockIdx.z (f32 atomic reduction) to keep ~2k waves in flight on small-N layers;
// when ksplit==1 the epilogue writes bias+acc (+ReLU) directly.
__global__ __launch_bounds__(WAVE)
void wmma_conv_kernel(const float* __restrict__ in, const float* __restrict__ w,
                      const float* __restrict__ bias, float* __restrict__ out,
                      int Cin, int H, int W, int Cout, int K,
                      int pad, int dil, int OH, int OW, int Kc, int kChunk,
                      int vecA, int direct, int relu) {
  int Npix = OH * OW;
  int KK   = K * K;
  int mBase  = blockIdx.x * 16;
  int nBase0 = blockIdx.y * 64;
  int ks = blockIdx.z * kChunk;
  int ke = ks + kChunk; if (ke > Kc) ke = Kc;
  if (ks >= Kc) return;                 // uniform across the wave

  int lane = threadIdx.x & 31;
  int hi   = lane >> 4;                 // lane group 0/1
  int lm   = lane & 15;

  int  mRow = mBase + lm;
  bool mOk  = mRow < Cout;
  const float* wrow = w + (size_t)(mOk ? mRow : 0) * Kc;
  float mMask = mOk ? 1.f : 0.f;

  v8f acc[4] = {};
  int  nColA[4]; bool nOkA[4]; int oyA[4], oxA[4];
#pragma unroll
  for (int t = 0; t < 4; ++t) {
    int nCol = nBase0 + 16 * t + lm;
    bool nOk = nCol < Npix;
    int  nCl = nOk ? nCol : 0;
    nColA[t] = nCol; nOkA[t] = nOk;
    oyA[t] = nCl / OW; oxA[t] = nCl - oyA[t] * OW;
  }

  for (int k0 = ks; k0 < ke; k0 += 32) {
    bool fullK = (k0 + 32 <= ke);       // uniform; false only on last step of last chunk
    v16h a;
    // ---- A fragment: lane<16 -> K {0..7,16..23}; lane>=16 -> K {8..15,24..31} ----
    if (vecA && fullK) {                // uniform branch; Kc%4==0 so rows 16B-aligned
      int r0 = k0 + 8 * hi;
      int r1 = k0 + 16 + 8 * hi;
      float4 w0 = *(const float4*)(wrow + r0);
      float4 w1 = *(const float4*)(wrow + r0 + 4);
      float4 w2 = *(const float4*)(wrow + r1);
      float4 w3 = *(const float4*)(wrow + r1 + 4);
      __builtin_prefetch(wrow + k0 + 128, 0, 3);   // global_prefetch_b8 weight stream
      a[0]=(_Float16)(w0.x*mMask); a[1]=(_Float16)(w0.y*mMask);
      a[2]=(_Float16)(w0.z*mMask); a[3]=(_Float16)(w0.w*mMask);
      a[4]=(_Float16)(w1.x*mMask); a[5]=(_Float16)(w1.y*mMask);
      a[6]=(_Float16)(w1.z*mMask); a[7]=(_Float16)(w1.w*mMask);
      a[8]=(_Float16)(w2.x*mMask); a[9]=(_Float16)(w2.y*mMask);
      a[10]=(_Float16)(w2.z*mMask); a[11]=(_Float16)(w2.w*mMask);
      a[12]=(_Float16)(w3.x*mMask); a[13]=(_Float16)(w3.y*mMask);
      a[14]=(_Float16)(w3.z*mMask); a[15]=(_Float16)(w3.w*mMask);
    } else {
#pragma unroll
      for (int e = 0; e < 16; ++e) {
        int kk = (e < 8 ? e : e + 8) + hi * 8;
        int k  = k0 + kk;
        bool ok = mOk && (k < ke);
        float v = wrow[ok ? k : 0];     // unconditional load, clamped address
        a[e] = (_Float16)(ok ? v : 0.f);
      }
    }
    // ---- 4 B fragments, one live at a time; lane<16 -> K {0..7,16..23}; lane>=16 -> K {8..15,24..31} ----
#pragma unroll
    for (int t = 0; t < 4; ++t) {
      v16h b;
#pragma unroll
      for (int e = 0; e < 16; ++e) {
        int k  = k0 + (e < 8 ? e : e + 8) + hi * 8;
        int ci = k / KK; int rr = k - ci * KK;
        int ky = rr / K, kx = rr - ky * K;
        int iy = oyA[t] + ky * dil - pad;
        int ix = oxA[t] + kx * dil - pad;
        bool ok = nOkA[t] && (fullK || (k < ke)) &&
                  (iy >= 0) && (iy < H) && (ix >= 0) && (ix < W);
        int off = ok ? (ci * H + iy) * W + ix : 0;
        float v = in[off];              // unconditional load, clamped address
        b[e] = (_Float16)(ok ? v : 0.f);
      }
      // Unconditional WMMA: invalid tiles have all-zero B -> acc unchanged.
      acc[t] = __builtin_amdgcn_wmma_f32_16x16x32_f16(false, a, false, b, (short)0,
                                                      acc[t], false, false);
    }
  }

  // C/D layout: VGPR r -> M = mBase + r + (hi?8:0), N = nBase + (lane&15)
#pragma unroll
  for (int t = 0; t < 4; ++t) {
#pragma unroll
    for (int r = 0; r < 8; ++r) {
      int m = mBase + r + hi * 8;
      if (m < Cout && nOkA[t]) {
        if (direct) {
          float v = bias[m] + acc[t][r];
          if (relu) v = fmaxf(v, 0.f);
          out[m * Npix + nColA[t]] = v;
        } else {
          atomicAdd(&out[m * Npix + nColA[t]], acc[t][r]);
        }
      }
    }
  }
}

// ---------------- 2x2 max pool ----------------
__global__ void maxpool2_kernel(const float* __restrict__ in, float* __restrict__ out,
                                int C, int H, int W, int OH, int OW) {
  int idx = blockIdx.x * blockDim.x + threadIdx.x;
  int total = C * OH * OW;
  if (idx >= total) return;
  int c = idx / (OH * OW);
  int r = idx - c * OH * OW;
  int oy = r / OW, ox = r - oy * OW;
  const float* p = in + (c * H + 2 * oy) * W + 2 * ox;
  out[idx] = fmaxf(fmaxf(p[0], p[1]), fmaxf(p[W], p[W + 1]));
}

// ---------------- bilinear (align-corners) 6->48 upsample + sigmoid, 21 channels ------
__global__ void upsample_sigmoid_kernel(const float* __restrict__ in, float* __restrict__ out) {
  int idx = blockIdx.x * blockDim.x + threadIdx.x;
  if (idx >= 21 * 48 * 48) return;
  int c = idx / 2304; int r = idx - c * 2304;
  int y = r / 48, x = r - y * 48;
  const float sc = 5.0f / 47.0f;
  float ty = y * sc, tx = x * sc;
  int y0 = (int)ty; int y1 = y0 + 1 < 5 ? y0 + 1 : 5; float fy = ty - (float)y0;
  int x0 = (int)tx; int x1 = x0 + 1 < 5 ? x0 + 1 : 5; float fx = tx - (float)x0;
  const float* p = in + c * 36;
  float v00 = p[y0 * 6 + x0], v01 = p[y0 * 6 + x1];
  float v10 = p[y1 * 6 + x0], v11 = p[y1 * 6 + x1];
  float vy0 = v00 * (1.f - fy) + v10 * fy;
  float vy1 = v01 * (1.f - fy) + v11 * fy;
  float v = vy0 * (1.f - fx) + vy1 * fx;
  out[idx] = 1.f / (1.f + expf(-v));
}

// ---------------- locally-connected layer: one wave per output element ----------------
// Streams local_w (504 MB, the model's bandwidth floor) as float2 (global_load_b64):
// rows are 10000 B (8-byte aligned), lanes read consecutive pairs -> fully coalesced.
__global__ __launch_bounds__(256)
void local_conn_kernel(const float* __restrict__ x, const float* __restrict__ lw,
                       const float* __restrict__ lb, float* __restrict__ out) {
  int wid  = blockIdx.x * (blockDim.x >> 5) + (threadIdx.x >> 5);
  int lane = threadIdx.x & 31;
  const int NOUT = 21 * 49 * 49;
  if (wid >= NOUT) return;              // uniform per wave
  int c = wid / 2401; int r = wid - c * 2401;
  int i = r / 49, j = r - i * 49;
  const float2* wr2 = (const float2*)(lw + (size_t)wid * 2500);
  const float*  xc  = x + c * 2304;
  float acc = 0.f;
  for (int t2 = lane; t2 < 1250; t2 += 32) {
    float2 wv = wr2[t2];
    __builtin_prefetch((const float*)(wr2 + t2 + 128), 0, 3);
    int t = t2 * 2;
    int k = t / 50, l = t - k * 50;     // t even and 50 even -> l, l+1 in same row
    int yy = i + k - 25;
    int xx = j + l - 25;
    bool yok = (yy >= 0) & (yy < 48);
    bool ok0 = yok & (xx >= 0) & (xx < 48);
    bool ok1 = yok & (xx + 1 >= 0) & (xx + 1 < 48);
    int base = yy * 48;
    float x0 = xc[ok0 ? base + xx     : 0];
    float x1 = xc[ok1 ? base + xx + 1 : 0];
    acc += (ok0 ? x0 : 0.f) * wv.x + (ok1 ? x1 : 0.f) * wv.y;
  }
  for (int off = 16; off > 0; off >>= 1) acc += __shfl_down(acc, off, 32);
  if (lane == 0) out[wid] = acc + lb[wid];
}

// ---------------- channel-mix linear: out[o,hw] = b[o] + sum_c w[o,c]*in[c,hw] --------
__global__ void channel_mix_kernel(const float* __restrict__ in, const float* __restrict__ w,
                                   const float* __restrict__ b, float* __restrict__ out,
                                   int Cin, int Cout, int Npix) {
  int idx = blockIdx.x * blockDim.x + threadIdx.x;
  if (idx >= Cout * Npix) return;
  int o = idx / Npix, hw = idx - o * Npix;
  float acc = b[o];
  for (int ci = 0; ci < Cin; ++ci) acc += w[o * Cin + ci] * in[ci * Npix + hw];
  out[idx] = acc;
}

// ---------------- min over 5 mixtures + channel softmax ----------------
__global__ void min_softmax_kernel(const float* __restrict__ in, float* __restrict__ out) {
  int hw = blockIdx.x * blockDim.x + threadIdx.x;
  if (hw >= 2401) return;
  float v[21];
#pragma unroll
  for (int c = 0; c < 21; ++c) {
    float m = in[c * 2401 + hw];
#pragma unroll
    for (int mix = 1; mix < 5; ++mix)
      m = fminf(m, in[(mix * 21 + c) * 2401 + hw]);
    v[c] = m;
  }
  float mx = v[0];
#pragma unroll
  for (int c = 1; c < 21; ++c) mx = fmaxf(mx, v[c]);
  float s = 0.f;
#pragma unroll
  for (int c = 0; c < 21; ++c) { v[c] = expf(v[c] - mx); s += v[c]; }
  float inv = 1.f / s;
#pragma unroll
  for (int c = 0; c < 21; ++c) out[c * 2401 + hw] = v[c] * inv;
}

// ---------------- host-side helper ----------------
static void launch_wmma_conv(const float* in, const float* w, const float* bias, float* out,
                             int Cin, int H, int W, int Cout, int K, int pad, int dil,
                             int OH, int OW, int relu, hipStream_t stream) {
  int Npix = OH * OW;
  int Kc   = Cin * K * K;
  int tot  = Cout * Npix;
  int mt  = (Cout + 15) / 16;
  int ntb = (Npix + 63) / 64;           // 4 N-tiles per wave
  int kIters = (Kc + 31) / 32;
  int ksplit = 1;
  while ((long)mt * ntb * ksplit < 2048 && ksplit * 2 <= kIters && ksplit < 64) ksplit <<= 1;
  int itersPer = (kIters + ksplit - 1) / ksplit;
  int kChunk   = itersPer * 32;
  int vecA   = ((Kc & 3) == 0) ? 1 : 0;
  int direct = (ksplit == 1) ? 1 : 0;
  if (!direct) fill_bias_kernel<<<(tot + 255) / 256, 256, 0, stream>>>(out, bias, Cout, Npix);
  dim3 grid(mt, ntb, ksplit);
  wmma_conv_kernel<<<grid, WAVE, 0, stream>>>(in, w, bias, out, Cin, H, W, Cout, K,
                                              pad, dil, OH, OW, Kc, kChunk,
                                              vecA, direct, relu);
  if (!direct && relu) relu_kernel<<<(tot + 255) / 256, 256, 0, stream>>>(out, tot);
}

extern "C" void kernel_launch(void* const* d_in, const int* in_sizes, int n_in,
                              void* d_out, int out_size, void* d_ws, size_t ws_size,
                              hipStream_t stream) {
  (void)in_sizes; (void)n_in; (void)out_size; (void)ws_size;
  const float* input = (const float*)d_in[0];
  const float* cw[16]; const float* cb[16];
  for (int i = 0; i < 16; ++i) { cw[i] = (const float*)d_in[1 + 2 * i]; cb[i] = (const float*)d_in[2 + 2 * i]; }
  const float* local_w = (const float*)d_in[33];
  const float* local_b = (const float*)d_in[34];
  const float* lin1_w  = (const float*)d_in[35];
  const float* lin1_b  = (const float*)d_in[36];
  const float* glob_w  = (const float*)d_in[37];
  const float* glob_b  = (const float*)d_in[38];
  const float* lin2_w  = (const float*)d_in[39];
  const float* lin2_b  = (const float*)d_in[40];
  float* out = (float*)d_out;

  float* ws = (float*)d_ws;
  const size_t BUF = 262144;            // floats per buffer (1 MB), 3 MB total scratch
  float* b0 = ws;
  float* b1 = ws + BUF;
  float* b2 = ws + 2 * BUF;

  static const int specs[16][3] = {
    {3,64,3},{64,64,3},{64,128,3},{128,128,3},{128,256,3},{256,256,3},{256,256,3},
    {256,512,3},{512,512,3},{512,512,3},{512,512,3},{512,512,3},{512,512,3},
    {512,4096,7},{4096,4096,1},{4096,21,1}};
  static const int cfg[16][2] = {
    {2,1},{2,1},{1,1},{1,1},{1,1},{1,1},{1,1},{1,1},{1,1},{1,1},
    {2,2},{2,2},{2,2},{12,4},{0,1},{0,1}};

  int H = 48;
  const float* cur = input;
  float* bufs[2] = {b0, b1};
  int pp = 0;
  for (int i = 0; i < 16; ++i) {
    int Cin = specs[i][0], Cout = specs[i][1], K = specs[i][2];
    int pad = cfg[i][0], dil = cfg[i][1];
    int OH = H + 2 * pad - dil * (K - 1);
    float* dst = bufs[pp];
    int relu = (i < 15) ? 1 : 0;
    if (i == 0 || i == 15) {
      int tot = Cout * OH * OH;
      conv_direct_kernel<<<(tot + 255) / 256, 256, 0, stream>>>(
          cur, cw[i], cb[i], dst, Cin, H, H, Cout, K, pad, dil, OH, OH, relu);
    } else {
      launch_wmma_conv(cur, cw[i], cb[i], dst, Cin, H, H, Cout, K, pad, dil, OH, OH, relu, stream);
    }
    cur = dst; pp ^= 1; H = OH;
    if (i == 1 || i == 3 || i == 6) {
      int PH = H / 2;
      float* pdst = bufs[pp];
      int tot = specs[i][1] * PH * PH;
      maxpool2_kernel<<<(tot + 255) / 256, 256, 0, stream>>>(cur, pdst, specs[i][1], H, H, PH, PH);
      cur = pdst; pp ^= 1; H = PH;
    }
  }
  // cur = conv15 output (21,6,6)
  float* ups = bufs[pp]; pp ^= 1;
  upsample_sigmoid_kernel<<<(21 * 48 * 48 + 255) / 256, 256, 0, stream>>>(cur, ups);

  // locally-connected layer: (21,48,48) -> (21,49,49), one wave per output
  {
    int nw = 21 * 49 * 49;
    int blocks = (nw + 7) / 8;          // 8 waves per 256-thread block
    local_conn_kernel<<<blocks, 256, 0, stream>>>(ups, local_w, local_b, b2);
  }

  // lin1 (21x21 channel mix)
  float* l1 = bufs[pp]; pp ^= 1;
  channel_mix_kernel<<<(21 * 2401 + 255) / 256, 256, 0, stream>>>(b2, lin1_w, lin1_b, l1, 21, 21, 2401);

  // glob conv 9x9 pad=4: (21,49,49) -> (105,49,49), WMMA GEMM, no relu
  float* g = bufs[pp]; pp ^= 1;
  launch_wmma_conv(l1, glob_w, glob_b, g, 21, 49, 49, 105, 9, 4, 1, 49, 49, 0, stream);

  // lin2 (105x105 channel mix)
  channel_mix_kernel<<<(105 * 2401 + 255) / 256, 256, 0, stream>>>(g, lin2_w, lin2_b, b2, 105, 105, 2401);

  // min over mixtures + channel softmax -> d_out (21,49,49)
  min_softmax_kernel<<<(2401 + 255) / 256, 256, 0, stream>>>(b2, out);
}
